// NeuralKernel_29652454212429
// MI455X (gfx1250) — matrix operand order repaired and checked by reading, not compiled, not run on hardware
//
#include <hip/hip_runtime.h>
#include <hip/hip_bf16.h>
#include <math.h>

typedef __bf16 bf16x8  __attribute__((ext_vector_type(8)));
typedef __bf16 bf16x16 __attribute__((ext_vector_type(16)));
typedef float  f32x4   __attribute__((ext_vector_type(4)));
typedef float  f32x8   __attribute__((ext_vector_type(8)));
typedef int    i32x4   __attribute__((ext_vector_type(4)));

#define NROWS   8192
#define KDIM    512
#define BT      128
#define KC      32
#define LDA     40            // padded LDS row stride (bf16); 80B rows, 16B aligned
#define SZ      (BT * LDA)    // bf16 elems per LDS tile array (5120)
#define PI_F    3.14159265358979f
#define EPS_F   1e-7f

#if __has_builtin(__builtin_amdgcn_global_load_async_to_lds_b128)
#define ASYNC_LDS 1
#else
#define ASYNC_LDS 0
#endif

typedef __attribute__((address_space(1))) i32x4* gbl_i32x4_p;
typedef __attribute__((address_space(3))) i32x4* lds_i32x4_p;

static __device__ __forceinline__ void copy16B_to_lds(const __bf16* g, __bf16* l) {
#if ASYNC_LDS
    __builtin_amdgcn_global_load_async_to_lds_b128(
        (gbl_i32x4_p)g, (lds_i32x4_p)l, 0, 0);
#else
    *(bf16x8*)l = *(const bf16x8*)g;
#endif
}

static __device__ __forceinline__ void wait_async_all() {
#if ASYNC_LDS
#if __has_builtin(__builtin_amdgcn_s_wait_asynccnt)
    __builtin_amdgcn_s_wait_asynccnt(0);
#else
    asm volatile("s_wait_asynccnt 0" ::: "memory");
#endif
#endif
}

static __device__ __forceinline__ bf16x16 frag_cat(bf16x8 a, bf16x8 b) {
    return __builtin_shufflevector(a, b, 0,1,2,3,4,5,6,7,8,9,10,11,12,13,14,15);
}

// Full 2-layer Erf-NTK elementwise recursion for one kernel entry.
static __device__ __forceinline__ float ntk_elem(float dot, float ax0, float ax1,
                                                 float ay0, float ay1) {
    const float inv_d = 1.0f / (float)KDIM;
    float nngp = dot * inv_d;
    float ntk  = nngp;
    float prod = ax0 * ay0;
    float r    = 2.0f * nngp * __frsqrt_rn(prod);
    r          = fminf(fmaxf(r, -1.0f + EPS_F), 1.0f - EPS_F);
    float nn   = (2.0f / PI_F) * asinf(r);
    float dth  = (4.0f / PI_F) * __frsqrt_rn(fmaxf(prod - 4.0f * nngp * nngp, EPS_F));
    ntk  = ntk * dth + nn;
    nngp = nn;
    prod = ax1 * ay1;
    r    = 2.0f * nngp * __frsqrt_rn(prod);
    r    = fminf(fmaxf(r, -1.0f + EPS_F), 1.0f - EPS_F);
    nn   = (2.0f / PI_F) * asinf(r);
    dth  = (4.0f / PI_F) * __frsqrt_rn(fmaxf(prod - 4.0f * nngp * nngp, EPS_F));
    ntk  = ntk * dth + nn;
    return ntk;
}

// One wave per row: c = |row|^2/d -> a0 = 1+2c, a1 = 1+2*erf_diag(c).
__global__ __launch_bounds__(256) void row_stats_kernel(
        const float* __restrict__ x, const float* __restrict__ y,
        float* __restrict__ ax0, float* __restrict__ ax1,
        float* __restrict__ ay0, float* __restrict__ ay1) {
    int row  = blockIdx.x * 8 + (threadIdx.x >> 5);
    int lane = threadIdx.x & 31;
    bool isY = row >= NROWS;
    int  r   = isY ? row - NROWS : row;
    const float* src = isY ? y : x;
    const f32x4* p = (const f32x4*)(src + (size_t)r * KDIM) + lane;
    float s = 0.0f;
#pragma unroll
    for (int q = 0; q < 4; ++q) {
        f32x4 v = p[32 * q];
        s += v[0]*v[0] + v[1]*v[1] + v[2]*v[2] + v[3]*v[3];
    }
#pragma unroll
    for (int off = 16; off; off >>= 1) s += __shfl_xor(s, off, 32);
    if (lane == 0) {
        float c  = s * (1.0f / (float)KDIM);
        float a0 = 1.0f + 2.0f * c;
        float cd = (2.0f / PI_F) * asinf(2.0f * c / (1.0f + 2.0f * c));
        float a1 = 1.0f + 2.0f * cd;
        if (isY) { ay0[r] = a0; ay1[r] = a1; }
        else     { ax0[r] = a0; ax1[r] = a1; }
    }
}

// One-time memory-bound prepass: f32 -> (hi, lo) bf16 split, 8 elems/thread.
__global__ __launch_bounds__(256) void split_bf16_kernel(
        const float* __restrict__ src, __bf16* __restrict__ hi,
        __bf16* __restrict__ lo, int n8) {
    int idx = blockIdx.x * 256 + threadIdx.x;
    if (idx >= n8) return;
    const f32x4* p = (const f32x4*)src + (size_t)idx * 2;
    f32x4 v0 = p[0], v1 = p[1];
    bf16x8 h, l;
#pragma unroll
    for (int q = 0; q < 4; ++q) {
        float v = v0[q]; __bf16 hh = (__bf16)v;
        h[q] = hh; l[q] = (__bf16)(v - (float)hh);
    }
#pragma unroll
    for (int q = 0; q < 4; ++q) {
        float v = v1[q]; __bf16 hh = (__bf16)v;
        h[4 + q] = hh; l[4 + q] = (__bf16)(v - (float)hh);
    }
    *(bf16x8*)(hi + (size_t)idx * 8) = h;
    *(bf16x8*)(lo + (size_t)idx * 8) = l;
}

// Fused bf16x3-split WMMA GEMM + NTK epilogue. 128x128 tile / workgroup,
// double-buffered LDS, async global->LDS staging when available.
__global__ __launch_bounds__(256, 1) void ntk_gemm_kernel(
        const __bf16* __restrict__ Xhi, const __bf16* __restrict__ Xlo,
        const __bf16* __restrict__ Yhi, const __bf16* __restrict__ Ylo,
        const float* __restrict__ ax0, const float* __restrict__ ax1,
        const float* __restrict__ ay0, const float* __restrict__ ay1,
        float* __restrict__ out) {
    __shared__ __align__(16) __bf16 sm[2 * 4 * SZ];   // [buf][Ahi,Alo,Bhi,Blo][BT*LDA]

    const int tile_i = blockIdx.y * BT;   // x rows
    const int tile_j = blockIdx.x * BT;   // y rows
    const int t    = threadIdx.x;
    const int lane = t & 31;
    const int w    = t >> 5;
    const int wr   = w >> 1;              // 0..3 : 32-row band
    const int wc   = w & 1;               // 0..1 : 64-col band

    f32x8 acc[2][4] = {};

    // staging: 2 threads/row, 16 bf16 (32B) per thread per array
    const int ldrow = t >> 1;
    const int kseg  = (t & 1) * 16;
    const size_t arow_off = (size_t)(tile_i + ldrow) * KDIM + kseg;
    const size_t brow_off = (size_t)(tile_j + ldrow) * KDIM + kseg;
    const int lds_off = ldrow * LDA + kseg;

    auto stage = [&](int k0, int p) {
        __bf16* base = sm + p * (4 * SZ) + lds_off;
        const __bf16* gAh = Xhi + arow_off + k0;
        const __bf16* gAl = Xlo + arow_off + k0;
        const __bf16* gBh = Yhi + brow_off + k0;
        const __bf16* gBl = Ylo + brow_off + k0;
        copy16B_to_lds(gAh,     base + 0 * SZ);
        copy16B_to_lds(gAh + 8, base + 0 * SZ + 8);
        copy16B_to_lds(gAl,     base + 1 * SZ);
        copy16B_to_lds(gAl + 8, base + 1 * SZ + 8);
        copy16B_to_lds(gBh,     base + 2 * SZ);
        copy16B_to_lds(gBh + 8, base + 2 * SZ + 8);
        copy16B_to_lds(gBl,     base + 3 * SZ);
        copy16B_to_lds(gBl + 8, base + 3 * SZ + 8);
    };

    stage(0, 0);
    wait_async_all();
    __syncthreads();

    const int arow = wr * 32 + (lane & 15);
    const int kha  = (lane >> 4) * 16;    // A frag: khalf in {0,16}
    const int brow = wc * 64 + (lane & 15);
    const int khb  = (lane >> 4) * 16;    // B frag: khalf in {0,16}

    int p = 0;
    for (int k0 = 0; k0 < KDIM; k0 += KC, p ^= 1) {
        const __bf16* sAhi = sm + p * (4 * SZ) + 0 * SZ;
        const __bf16* sAlo = sm + p * (4 * SZ) + 1 * SZ;
        const __bf16* sBhi = sm + p * (4 * SZ) + 2 * SZ;
        const __bf16* sBlo = sm + p * (4 * SZ) + 3 * SZ;

        // ---- fragment loads first (their dscnt wait excludes later stores)
        bf16x16 fAh[2], fAl[2];
#pragma unroll
        for (int mt = 0; mt < 2; ++mt) {
            const __bf16* ph = sAhi + (arow + mt * 16) * LDA + kha;
            const __bf16* pl = sAlo + (arow + mt * 16) * LDA + kha;
            fAh[mt] = frag_cat(*(const bf16x8*)ph, *(const bf16x8*)(ph + 8));
            fAl[mt] = frag_cat(*(const bf16x8*)pl, *(const bf16x8*)(pl + 8));
        }
        bf16x16 fBh[4], fBl[4];
#pragma unroll
        for (int nt = 0; nt < 4; ++nt) {
            const __bf16* ph = sBhi + (brow + nt * 16) * LDA + khb;
            const __bf16* pl = sBlo + (brow + nt * 16) * LDA + khb;
            fBh[nt] = frag_cat(*(const bf16x8*)ph, *(const bf16x8*)(ph + 8));
            fBl[nt] = frag_cat(*(const bf16x8*)pl, *(const bf16x8*)(pl + 8));
        }

        // ---- prefetch next chunk into the other buffer (overlaps WMMAs)
        if (k0 + KC < KDIM) stage(k0 + KC, p ^ 1);

        // ---- bf16x3 split products: hi*hi + hi*lo + lo*hi
#pragma unroll
        for (int mt = 0; mt < 2; ++mt) {
#pragma unroll
            for (int nt = 0; nt < 4; ++nt) {
                acc[mt][nt] = __builtin_amdgcn_wmma_f32_16x16x32_bf16(
                    false, fAh[mt], false, fBh[nt], (short)0, acc[mt][nt], false, false);
                acc[mt][nt] = __builtin_amdgcn_wmma_f32_16x16x32_bf16(
                    false, fAh[mt], false, fBl[nt], (short)0, acc[mt][nt], false, false);
                acc[mt][nt] = __builtin_amdgcn_wmma_f32_16x16x32_bf16(
                    false, fAl[mt], false, fBh[nt], (short)0, acc[mt][nt], false, false);
            }
        }
        wait_async_all();
        __syncthreads();
    }

    // ---- fused NTK epilogue; C/D layout: lane = N, VGPR r -> M = r + 8*(lane>=16)
    const int col   = lane & 15;
    const int rhalf = (lane >> 4) * 8;
    float a0v[2][8], a1v[2][8];
#pragma unroll
    for (int mt = 0; mt < 2; ++mt) {
        const int ibase = tile_i + wr * 32 + mt * 16 + rhalf;
#pragma unroll
        for (int r = 0; r < 8; ++r) {
            a0v[mt][r] = ax0[ibase + r];
            a1v[mt][r] = ax1[ibase + r];
        }
    }
    const int jbase = tile_j + wc * 64;
#pragma unroll
    for (int nt = 0; nt < 4; ++nt) {
        const int j = jbase + nt * 16 + col;
        const float y0 = ay0[j], y1 = ay1[j];
#pragma unroll
        for (int mt = 0; mt < 2; ++mt) {
            const int ibase = tile_i + wr * 32 + mt * 16 + rhalf;
#pragma unroll
            for (int r = 0; r < 8; ++r) {
                out[(size_t)(ibase + r) * NROWS + j] =
                    ntk_elem(acc[mt][nt][r], a0v[mt][r], a1v[mt][r], y0, y1);
            }
        }
    }
}

extern "C" void kernel_launch(void* const* d_in, const int* in_sizes, int n_in,
                              void* d_out, int out_size, void* d_ws, size_t ws_size,
                              hipStream_t stream) {
    const float* x = (const float*)d_in[0];
    const float* y = (const float*)d_in[1];
    float* ws  = (float*)d_ws;
    // ws layout: 4*8192 f32 row stats (128 KB), then 4 bf16 matrices (32 MB)
    float* ax0 = ws;
    float* ax1 = ws + NROWS;
    float* ay0 = ws + 2 * NROWS;
    float* ay1 = ws + 3 * NROWS;
    __bf16* bufs = (__bf16*)(ws + 4 * NROWS);
    __bf16* Xhi = bufs;
    __bf16* Xlo = bufs + (size_t)NROWS * KDIM;
    __bf16* Yhi = bufs + (size_t)2 * NROWS * KDIM;
    __bf16* Ylo = bufs + (size_t)3 * NROWS * KDIM;

    row_stats_kernel<<<(2 * NROWS) / 8, 256, 0, stream>>>(x, y, ax0, ax1, ay0, ay1);

    const int n8 = NROWS * KDIM / 8;
    split_bf16_kernel<<<n8 / 256, 256, 0, stream>>>(x, Xhi, Xlo, n8);
    split_bf16_kernel<<<n8 / 256, 256, 0, stream>>>(y, Yhi, Ylo, n8);

    dim3 grid(NROWS / BT, NROWS / BT);
    ntk_gemm_kernel<<<grid, 256, 0, stream>>>(Xhi, Xlo, Yhi, Ylo,
                                              ax0, ax1, ay0, ay1, (float*)d_out);
}
